// MyMSA_9371618640658
// MI455X (gfx1250) — compile-verified, matrix operand order checked by reading, not run on hardware
//
#include <hip/hip_runtime.h>

// ---------------------------------------------------------------------------
// MI455X / gfx1250, wave32. Low-rank QKV projection + flash attention.
// All matmuls via V_WMMA_F32_16X16X32_F16 (f16 in, f32 accum).
// Weights and K/V tiles staged in LDS and shared across the 8 waves of each
// workgroup (barrier-synced), with register double-buffering of the K/V
// stream in the attention loop.
// ---------------------------------------------------------------------------

typedef _Float16 f16;
typedef __attribute__((ext_vector_type(16))) _Float16 v16h;
typedef __attribute__((ext_vector_type(8)))  _Float16 v8h;
typedef __attribute__((ext_vector_type(8)))  float    v8f;

#define WMMA16x16x32(a, b, c) \
  __builtin_amdgcn_wmma_f32_16x16x32_f16(false, (a), false, (b), (short)0, (c), false, false)

constexpr int kB = 8, kS = 1024, kD = 1024, kH = 8, kR = 32, kDH = 128;
constexpr int PWAVES = 8;               // projection waves per block (256 thr)
constexpr int AWAVES = 8;               // attention waves per block (256 thr)

__device__ __forceinline__ v8f vzero8() {
  v8f z;
#pragma unroll
  for (int i = 0; i < 8; ++i) z[i] = 0.0f;
  return z;
}

// A-fragment (16x32 f16, M x K) load from row-major f16 memory (global or LDS).
// Per ISA 7.12.2: lane L holds row m = L&15; K chunks [k0, k0+8) and
// [k0+16, k0+24) where k0 = (L>=16)*8.  Same pattern serves the B operand
// when the source is the *transposed* B matrix stored N x K row-major.
__device__ __forceinline__ v16h load_frag_f16(const f16* base, int ld) {
  const int lane = threadIdx.x & 31;
  const int row  = lane & 15;
  const int k0   = (lane >> 4) << 3;
  const f16* p = base + (size_t)row * ld + k0;
  const v8h lo = *(const v8h*)(p);
  const v8h hi = *(const v8h*)(p + 16);
  v16h r;
#pragma unroll
  for (int i = 0; i < 8; ++i) { r[i] = lo[i]; r[i + 8] = hi[i]; }
  return r;
}

// Same fragment pattern but converting from f32 memory on the fly (x only).
__device__ __forceinline__ v16h load_frag_f32(const float* base, int ld) {
  const int lane = threadIdx.x & 31;
  const int row  = lane & 15;
  const int k0   = (lane >> 4) << 3;
  const float* p = base + (size_t)row * ld + k0;
  v16h r;
#pragma unroll
  for (int i = 0; i < 8; ++i) { r[i] = (f16)p[i]; r[i + 8] = (f16)p[i + 16]; }
  return r;
}

// ---------------------------------------------------------------------------
// Projection kernel: y = (x_h . V^T) . U^T + b for q, k, v of one head.
// Per-head weights converted to f16 once per workgroup into LDS.
// ---------------------------------------------------------------------------

// Cooperative f32 -> f16 staging of one head's V [R x DH] and U [DH x R].
__device__ __forceinline__ void stage_weights(const float* __restrict__ U,
                                              const float* __restrict__ Vm,
                                              int h, f16* ldsU, f16* ldsV) {
  const int tid   = threadIdx.x;
  const int nthr  = PWAVES * 32;
  const float* vs = Vm + (size_t)h * kR * kDH;     // 4096 f32, contiguous
  const float* us = U  + (size_t)h * kDH * kR;     // 4096 f32, contiguous
#pragma unroll
  for (int base = 0; base < kR * kDH; base += nthr * 8) {
    const int off = base + tid * 8;
#pragma unroll
    for (int e = 0; e < 8; ++e) {
      ldsV[off + e] = (f16)vs[off + e];
      ldsU[off + e] = (f16)us[off + e];
    }
  }
}

// One low-rank projection for this wave's 16-token tile, weights from LDS.
__device__ __forceinline__ void lowrank_proj(const v16h xa[4],
                                             const f16* ldsU, const f16* ldsV,
                                             const float* __restrict__ bias,
                                             int h, f16* ldsT, v8f yacc[8]) {
  const int lane  = threadIdx.x & 31;
  const int n     = lane & 15;
  const int mbase = (lane >> 4) * 8;

  // t = x_h . V^T : N = R = 32 (2 tiles), K = DH = 128 (4 chunks)
  v8f tacc[2];
  tacc[0] = vzero8(); tacc[1] = vzero8();
#pragma unroll
  for (int nt = 0; nt < 2; ++nt)
#pragma unroll
    for (int c = 0; c < 4; ++c) {
      v16h vb = load_frag_f16(ldsV + (nt * 16) * kDH + c * 32, kDH);
      tacc[nt] = WMMA16x16x32(xa[c], vb, tacc[nt]);
    }

  // C/D layout -> wave-private LDS (16x32 f16) -> A fragment
#pragma unroll
  for (int nt = 0; nt < 2; ++nt)
#pragma unroll
    for (int r = 0; r < 8; ++r)
      ldsT[(mbase + r) * kR + nt * 16 + n] = (f16)tacc[nt][r];
  v16h tfrag = load_frag_f16(ldsT, kR);

  // y = t . U^T + b : N = DH = 128 (8 tiles), K = R = 32 (1 chunk)
#pragma unroll
  for (int d = 0; d < 8; ++d) {
    v16h ub = load_frag_f16(ldsU + (d * 16) * kR, kR);
    v8f acc = WMMA16x16x32(tfrag, ub, vzero8());
    const float bias_n = bias[h * kDH + d * 16 + n];
#pragma unroll
    for (int r = 0; r < 8; ++r) acc[r] += bias_n;
    yacc[d] = acc;
  }
}

__global__ __launch_bounds__(PWAVES * 32) void qkv_lowrank_kernel(
    const float* __restrict__ x,
    const float* __restrict__ Uq, const float* __restrict__ Vq, const float* __restrict__ bq,
    const float* __restrict__ Uk, const float* __restrict__ Vk, const float* __restrict__ bk,
    const float* __restrict__ Uv, const float* __restrict__ Vv, const float* __restrict__ bv,
    f16* __restrict__ qf, f16* __restrict__ kf, f16* __restrict__ vt) {
  __shared__ __align__(16) f16 ldsV[kR * kDH];            // 8KB
  __shared__ __align__(16) f16 ldsU[kDH * kR];            // 8KB
  __shared__ __align__(16) f16 ldsT[PWAVES][16 * kR];     // 8KB, wave-private t
  const int wave  = threadIdx.x >> 5;
  const int lane  = threadIdx.x & 31;
  const int n     = lane & 15;
  const int mbase = (lane >> 4) * 8;
  const int h     = blockIdx.y;
  const int tok0  = (blockIdx.x * PWAVES + wave) * 16;    // [0, B*S)
  const int bb    = tok0 / kS;
  const int s0    = tok0 % kS;

  // x_h tile as 4 A-fragments over K = DH (f32 -> f16 on the fly)
  v16h xa[4];
#pragma unroll
  for (int c = 0; c < 4; ++c)
    xa[c] = load_frag_f32(x + (size_t)tok0 * kD + h * kDH + c * 32, kD);

  v8f y[8];
  // ---- Q : f16 [B,H,S,DH] ----
  stage_weights(Uq, Vq, h, ldsU, ldsV);
  __syncthreads();
  lowrank_proj(xa, ldsU, ldsV, bq, h, ldsT[wave], y);
  {
    f16* qb = qf + (((size_t)bb * kH + h) * kS + s0) * kDH;
#pragma unroll
    for (int d = 0; d < 8; ++d)
#pragma unroll
      for (int r = 0; r < 8; ++r)
        qb[(size_t)(mbase + r) * kDH + d * 16 + n] = (f16)y[d][r];
  }
  // ---- K : f16 [B,H,S,DH] ----
  __syncthreads();
  stage_weights(Uk, Vk, h, ldsU, ldsV);
  __syncthreads();
  lowrank_proj(xa, ldsU, ldsV, bk, h, ldsT[wave], y);
  {
    f16* kb = kf + (((size_t)bb * kH + h) * kS + s0) * kDH;
#pragma unroll
    for (int d = 0; d < 8; ++d)
#pragma unroll
      for (int r = 0; r < 8; ++r)
        kb[(size_t)(mbase + r) * kDH + d * 16 + n] = (f16)y[d][r];
  }
  // ---- V : stored transposed, f16 [B,H,DH,S] (PV's B operand N x K) ----
  __syncthreads();
  stage_weights(Uv, Vv, h, ldsU, ldsV);
  __syncthreads();
  lowrank_proj(xa, ldsU, ldsV, bv, h, ldsT[wave], y);
  {
    f16* vb_ = vt + ((size_t)bb * kH + h) * (size_t)kDH * kS;
#pragma unroll
    for (int d = 0; d < 8; ++d)
#pragma unroll
      for (int r = 0; r < 8; ++r)
        vb_[(size_t)(d * 16 + n) * kS + s0 + mbase + r] = (f16)y[d][r];
  }
}

// ---------------------------------------------------------------------------
// Flash attention: 8 waves x 16 queries per block, one (b,h) per blockIdx.y.
// K/V tiles (32 keys) staged in LDS shared by all waves; next tile prefetched
// into registers while WMMAs consume the current one.
// ---------------------------------------------------------------------------

__device__ __forceinline__ void load_kv_regs(const f16* __restrict__ Kb,
                                             const f16* __restrict__ Vb,
                                             int j0, int tid,
                                             v8h kreg[2], v8h vreg[2]) {
#pragma unroll
  for (int i = 0; i < 2; ++i) {
    const int cid = tid + i * (AWAVES * 32);        // 0..511, 16B chunks
    // K tile: 32 rows x 128 halves, contiguous 8KB block at Kb + j0*DH
    kreg[i] = *(const v8h*)(Kb + (size_t)j0 * kDH + cid * 8);
    // V^T tile: 128 rows (d) x 32 halves (keys), row stride S
    const int row = cid >> 2, c = cid & 3;
    vreg[i] = *(const v8h*)(Vb + (size_t)row * kS + j0 + c * 8);
  }
}

__global__ __launch_bounds__(AWAVES * 32) void attn_kernel(
    const f16* __restrict__ qf, const f16* __restrict__ kf,
    const f16* __restrict__ vt, float* __restrict__ out) {
  __shared__ __align__(16) f16 ldsK[32 * kDH];            // 8KB [key][dh]
  __shared__ __align__(16) f16 ldsVt[kDH * 32];           // 8KB [d][key]
  __shared__ __align__(16) f16 ldsP[AWAVES][16 * 32];     // 8KB, wave-private P
  const int tid   = threadIdx.x;
  const int wave  = tid >> 5;
  const int lane  = tid & 31;
  const int n     = lane & 15;
  const int mbase = (lane >> 4) * 8;
  const int bh    = blockIdx.y;                           // b*H + h
  const int bb    = bh / kH, h = bh % kH;
  const int q0    = (blockIdx.x * AWAVES + wave) * 16;

  const f16* Qb = qf + ((size_t)bh * kS + q0) * kDH;
  const f16* Kb = kf + (size_t)bh * kS * kDH;
  const f16* Vb = vt + (size_t)bh * (size_t)kDH * kS;

  v16h qa[4];
#pragma unroll
  for (int c = 0; c < 4; ++c) qa[c] = load_frag_f16(Qb + c * 32, kDH);

  v8f oacc[8];
#pragma unroll
  for (int d = 0; d < 8; ++d) oacc[d] = vzero8();
  float mrow[8], lrow[8];
#pragma unroll
  for (int r = 0; r < 8; ++r) { mrow[r] = -1e30f; lrow[r] = 0.0f; }

  const float scale = 0.08838834764831845f;               // 1/sqrt(DH)

  v8h kreg[2], vreg[2];
  load_kv_regs(Kb, Vb, 0, tid, kreg, vreg);               // prologue

  for (int j0 = 0; j0 < kS; j0 += 32) {
    __syncthreads();                                      // LDS free to rewrite
#pragma unroll
    for (int i = 0; i < 2; ++i) {
      const int cid = tid + i * (AWAVES * 32);
      *(v8h*)(ldsK + cid * 8) = kreg[i];
      const int row = cid >> 2, c = cid & 3;
      *(v8h*)(ldsVt + row * 32 + c * 8) = vreg[i];
    }
    __syncthreads();                                      // tile visible
    if (j0 + 32 < kS)                                     // uniform branch
      load_kv_regs(Kb, Vb, j0 + 32, tid, kreg, vreg);     // overlap next tile

    // S tile: Q(16x128) . K^T -> two 16x16 f32 fragments (32 keys)
    v8f sfr[2];
    sfr[0] = vzero8(); sfr[1] = vzero8();
#pragma unroll
    for (int kt = 0; kt < 2; ++kt)
#pragma unroll
      for (int c = 0; c < 4; ++c) {
        v16h kb = load_frag_f16(ldsK + (kt * 16) * kDH + c * 32, kDH);
        sfr[kt] = WMMA16x16x32(qa[c], kb, sfr[kt]);
      }

#pragma unroll
    for (int kt = 0; kt < 2; ++kt)
#pragma unroll
      for (int r = 0; r < 8; ++r) sfr[kt][r] *= scale;

    // per-row max across 32 keys: local pair-max then 16-lane butterfly
    float mx[8];
#pragma unroll
    for (int r = 0; r < 8; ++r) mx[r] = fmaxf(sfr[0][r], sfr[1][r]);
#pragma unroll
    for (int off = 1; off <= 8; off <<= 1)
#pragma unroll
      for (int r = 0; r < 8; ++r) mx[r] = fmaxf(mx[r], __shfl_xor(mx[r], off, 32));

    // online softmax update
    float rs[8];
#pragma unroll
    for (int r = 0; r < 8; ++r) {
      const float mn   = fmaxf(mrow[r], mx[r]);
      const float corr = __expf(mrow[r] - mn);
      mrow[r] = mn;
      const float p0 = __expf(sfr[0][r] - mn);
      const float p1 = __expf(sfr[1][r] - mn);
      sfr[0][r] = p0; sfr[1][r] = p1;
      rs[r] = p0 + p1;
      lrow[r] *= corr;
#pragma unroll
      for (int d = 0; d < 8; ++d) oacc[d][r] *= corr;
    }
#pragma unroll
    for (int off = 1; off <= 8; off <<= 1)
#pragma unroll
      for (int r = 0; r < 8; ++r) rs[r] += __shfl_xor(rs[r], off, 32);
#pragma unroll
    for (int r = 0; r < 8; ++r) lrow[r] += rs[r];

    // P (C/D layout) -> wave-private LDS -> A fragment, then O += P . V
    f16* Pl = ldsP[wave];
#pragma unroll
    for (int kt = 0; kt < 2; ++kt)
#pragma unroll
      for (int r = 0; r < 8; ++r)
        Pl[(mbase + r) * 32 + kt * 16 + n] = (f16)sfr[kt][r];
    v16h pfrag = load_frag_f16(Pl, 32);
#pragma unroll
    for (int d = 0; d < 8; ++d) {
      v16h vbf = load_frag_f16(ldsVt + (d * 16) * 32, 32);
      oacc[d] = WMMA16x16x32(pfrag, vbf, oacc[d]);
    }
  }

  // epilogue: normalize and scatter fp32 output [B,S,D]
  float inv[8];
#pragma unroll
  for (int r = 0; r < 8; ++r) inv[r] = 1.0f / lrow[r];
  float* ob = out + ((size_t)bb * kS + q0) * kD + h * kDH;
#pragma unroll
  for (int d = 0; d < 8; ++d)
#pragma unroll
    for (int r = 0; r < 8; ++r)
      ob[(size_t)(mbase + r) * kD + d * 16 + n] = oacc[d][r] * inv[r];
}

extern "C" void kernel_launch(void* const* d_in, const int* in_sizes, int n_in,
                              void* d_out, int out_size, void* d_ws, size_t ws_size,
                              hipStream_t stream) {
  (void)in_sizes; (void)n_in; (void)out_size; (void)ws_size;
  const float* x  = (const float*)d_in[0];
  const float* Uq = (const float*)d_in[1];
  const float* Vq = (const float*)d_in[2];
  const float* bq = (const float*)d_in[3];
  const float* Uk = (const float*)d_in[4];
  const float* Vk = (const float*)d_in[5];
  const float* bk = (const float*)d_in[6];
  const float* Uv = (const float*)d_in[7];
  const float* Vv = (const float*)d_in[8];
  const float* bv = (const float*)d_in[9];
  float* out = (float*)d_out;

  // f16 staging in workspace: Q [B,H,S,DH], K [B,H,S,DH], V^T [B,H,DH,S]
  f16* qf = (f16*)d_ws;
  f16* kf = qf + (size_t)kB * kH * kS * kDH;
  f16* vt = kf + (size_t)kB * kH * kS * kDH;

  dim3 pgrid((kB * kS / 16) / PWAVES, kH);
  qkv_lowrank_kernel<<<pgrid, PWAVES * 32, 0, stream>>>(
      x, Uq, Vq, bq, Uk, Vk, bk, Uv, Vv, bv, qf, kf, vt);

  dim3 agrid((kS / 16) / AWAVES, kB * kH);
  attn_kernel<<<agrid, AWAVES * 32, 0, stream>>>(qf, kf, vt, out);
}
